// RelativeAttention_31482110280308
// MI455X (gfx1250) — matrix operand order repaired and checked by reading, not compiled, not run on hardware
//
#include <hip/hip_runtime.h>
#include <hip/hip_bf16.h>

// ---------------------------------------------------------------------------
// Relative-position attention, fused flash-style, bf16 WMMA (gfx1250 / MI455X)
//   N=4, L=1024, D_MODEL=1024, H=16, depth=64, MAX_SEQ=2048
// ---------------------------------------------------------------------------

typedef __bf16 bf16_t;
typedef __attribute__((ext_vector_type(16))) bf16_t v16bf;
typedef __attribute__((ext_vector_type(8)))  float  v8f;
typedef __attribute__((ext_vector_type(4)))  int    v4i;

#define N_B    4
#define L_SEQ  1024
#define DM     1024
#define NH     16
#define DH     64
#define MAXSEQ 2048
#define EPAD   (L_SEQ + 32)   // E padded so QE band tiles can over-read zeros

#define AS1 __attribute__((address_space(1)))
#define AS3 __attribute__((address_space(3)))

#if defined(__gfx1250__) && __has_builtin(__builtin_amdgcn_global_load_async_to_lds_b128)
#define HAVE_ASYNC_LDS 1
#else
#define HAVE_ASYNC_LDS 0
#endif

static __device__ __forceinline__ v8f wmma_bf16(v16bf a, v16bf b, v8f c) {
    // D = A(16x32 bf16) * B(32x16 bf16) + C(16x16 f32)
    return __builtin_amdgcn_wmma_f32_16x16x32_bf16(
        /*neg_a=*/false, a, /*neg_b=*/false, b,
        /*c_mod=*/(short)0, c, /*reuse_a=*/false, /*reuse_b=*/false);
}

#if HAVE_ASYNC_LDS
// generic shared pointer -> LDS (addrspace 3): flat aperture maps addr[31:0]
static __device__ __forceinline__ AS3 v4i* lds_v4i(void* p) {
    return (AS3 v4i*)(unsigned int)(unsigned long long)p;
}
static __device__ __forceinline__ AS1 v4i* gbl_v4i(const void* p) {
    return (AS1 v4i*)(unsigned long long)p;
}
#endif

static __device__ __forceinline__ void wait_async0() {
#if defined(__gfx1250__)
#if __has_builtin(__builtin_amdgcn_s_wait_asynccnt)
    __builtin_amdgcn_s_wait_asynccnt(0);
#else
    asm volatile("s_wait_asynccnt 0x0" ::: "memory");
#endif
#endif
}

// ---- weight transpose + f32->bf16: Wt[n][k] = W[k][n] --------------------
__global__ __launch_bounds__(256)
void k_transpose_w(const float* __restrict__ W, bf16_t* __restrict__ Wt) {
    int idx = blockIdx.x * 256 + threadIdx.x;       // 1M elements exactly
    int nn = idx >> 10, kk = idx & 1023;
    Wt[idx] = (bf16_t)W[kk * DM + nn];
}

// ---- build padded bf16 E: E[m][d] = pos_emb[MAXSEQ-L+m][d], zeros beyond --
__global__ __launch_bounds__(256)
void k_make_E(const float* __restrict__ pos, bf16_t* __restrict__ E) {
    int idx = blockIdx.x * 256 + threadIdx.x;       // EPAD*64 exactly
    int m = idx >> 6, d = idx & 63;
    float v = (m < L_SEQ) ? pos[(MAXSEQ - L_SEQ + m) * DH + d] : 0.0f;
    E[idx] = (bf16_t)v;
}

// ---- [4096x1024] x [1024x1024] GEMM, 128x128 block tile, 8 waves ----------
// Wave w owns M rows [bm*128 + w*16, +16) x all 128 N columns (8 accumulators).
// The shared B panel (128 N x 32 K bf16, 8 KB) is double-buffered in LDS and
// streamed with async global->LDS loads (ASYNCcnt) when available.
// MODE 0: out = bf16 head-split  [n][h][l][d]           (Q with scale=1/8, K)
// MODE 1: out = bf16 head-transposed [n][h][d][l]       (V)
// MODE 2: out = f32 row-major [row][col]                (final O projection)
template <typename AT, int MODE>
__global__ __launch_bounds__(256)
void k_gemm(const AT* __restrict__ A, const bf16_t* __restrict__ Bt,
            const float* __restrict__ bias, void* __restrict__ Out,
            float scale) {
    __shared__ bf16_t Bsh[2][128 * 32];              // [buf][n*32 + k]

    const int tid  = threadIdx.x;
    const int lane = tid & 31, wave = tid >> 5;
    const int bm = blockIdx.x >> 3, bn = blockIdx.x & 7;   // 32 x 8 blocks
    const int half = lane >> 4, colj = lane & 15;
    const int kb = half * 8, ks = half * 16;

    const AT*     ap     = A  + (size_t)(bm * 128 + wave * 16 + colj) * DM;
    const bf16_t* btBase = Bt + (size_t)(bn * 128) * DM;

    // staging map: thread -> rows {sn, sn+64}, 16-byte chunk sc (8 bf16)
    const int sn = tid >> 2, sc = tid & 3;

    auto stage = [&](int buf, int k0) {
        const bf16_t* g0 = btBase + (size_t)sn * DM + k0 + sc * 8;
        const bf16_t* g1 = btBase + (size_t)(sn + 64) * DM + k0 + sc * 8;
        bf16_t* l0 = &Bsh[buf][sn * 32 + sc * 8];
        bf16_t* l1 = &Bsh[buf][(sn + 64) * 32 + sc * 8];
#if HAVE_ASYNC_LDS
        __builtin_amdgcn_global_load_async_to_lds_b128(gbl_v4i(g0), lds_v4i(l0), 0, 0);
        __builtin_amdgcn_global_load_async_to_lds_b128(gbl_v4i(g1), lds_v4i(l1), 0, 0);
#else
        uint4 d0 = *(const uint4*)g0;
        uint4 d1 = *(const uint4*)g1;
        *(uint4*)l0 = d0;
        *(uint4*)l1 = d1;
#endif
    };

    v8f acc[8];
    {
        v8f z = {};
#pragma unroll
        for (int j = 0; j < 8; ++j) acc[j] = z;
    }

    stage(0, 0);
    for (int k0 = 0; k0 < DM; k0 += 32) {
        const int buf = (k0 >> 5) & 1;
        wait_async0();            // my panel writes for `buf` are complete
        __syncthreads();          // everyone's are; prev buffer fully consumed
        if (k0 + 32 < DM) stage(buf ^ 1, k0 + 32);

        // A fragment (wave-private rows, converted f32->bf16 on the fly)
        v16bf a;
#pragma unroll
        for (int j = 0; j < 8; ++j) {
            a[j]     = (bf16_t)ap[k0 + ks + j];
            a[8 + j] = (bf16_t)ap[k0 + ks + 8 + j];
        }
        // 8 WMMAs against the shared B panel
        const bf16_t* bsh = &Bsh[buf][0];
#pragma unroll
        for (int j = 0; j < 8; ++j) {
            const bf16_t* bp = bsh + (j * 16 + colj) * 32 + ks;
            v16bf b;
#pragma unroll
            for (int e = 0; e < 16; ++e) b[e] = bp[e];
            acc[j] = wmma_bf16(a, b, acc[j]);
        }
    }

#pragma unroll
    for (int j = 0; j < 8; ++j) {
        const int col = bn * 128 + j * 16 + colj;
#pragma unroll
        for (int r = 0; r < 8; ++r) {
            const int row = bm * 128 + wave * 16 + r + 8 * half;
            const float v = (acc[j][r] + bias[col]) * scale;
            if (MODE == 2) {
                ((float*)Out)[(size_t)row * DM + col] = v;
            } else {
                const int n = row >> 10, l = row & 1023;
                const int h = col >> 6,  d = col & 63;
                if (MODE == 0)
                    ((bf16_t*)Out)[(((size_t)(n * NH + h)) * L_SEQ + l) * DH + d] = (bf16_t)v;
                else
                    ((bf16_t*)Out)[(((size_t)(n * NH + h)) * DH + d) * L_SEQ + l] = (bf16_t)v;
            }
        }
    }
}

// ---- fused causal attention + relative bias, one wave per 16-row block ----
__global__ __launch_bounds__(256)
void k_attn(const bf16_t* __restrict__ Qb, const bf16_t* __restrict__ Kb,
            const bf16_t* __restrict__ Vt, const bf16_t* __restrict__ E,
            bf16_t* __restrict__ Ob) {
    __shared__ bf16_t pbuf[8][16 * 32];              // per-wave P transpose

    const int lane = threadIdx.x & 31, wave = threadIdx.x >> 5;
    const int w = blockIdx.x * 8 + wave;             // 4096 waves total
    const int n = w >> 10, rem = w & 1023;
    const int h = rem >> 6, l0 = (rem & 63) * 16;
    const int half = lane >> 4, colj = lane & 15;
    const int kb = half * 8, ks = half * 16;

    const size_t hoff = (size_t)(n * NH + h);
    const bf16_t* Qh = Qb + hoff * L_SEQ * DH;
    const bf16_t* Kh = Kb + hoff * L_SEQ * DH;
    const bf16_t* Vh = Vt + hoff * DH * L_SEQ;       // [d][l]
    bf16_t* pw = &pbuf[wave][0];

    // Q fragments for the whole block (K = depth 64 -> two 16x32 frags)
    v16bf aq0, aq1;
    {
        const bf16_t* qp = Qh + (size_t)(l0 + colj) * DH;
#pragma unroll
        for (int j = 0; j < 8; ++j) {
            aq0[j]     = qp[ks + j];       aq0[8 + j] = qp[ks + 8 + j];
            aq1[j]     = qp[32 + ks + j];  aq1[8 + j] = qp[32 + ks + 8 + j];
        }
    }

    float rowmax[8], rowsum[8];
#pragma unroll
    for (int r = 0; r < 8; ++r) { rowmax[r] = -3.0e38f; rowsum[r] = 0.0f; }
    v8f o0 = {}, o1 = {}, o2 = {}, o3 = {};

    for (int m0 = 0; m0 <= l0; m0 += 32) {
        // prefetch next key block while we compute this one
        __builtin_prefetch(Kh + (size_t)(m0 + 32 + colj) * DH, 0, 0);

        // ---- S = Q K^T (two 16-col tiles, K-dim = 64) ----
        v8f sa = {}, sb = {};
        {
            const bf16_t* kp = Kh + (size_t)(m0 + colj) * DH;
            v16bf b0, b1;
#pragma unroll
            for (int j = 0; j < 16; ++j) { b0[j] = kp[ks + j]; b1[j] = kp[32 + ks + j]; }
            sa = wmma_bf16(aq0, b0, sa);  sa = wmma_bf16(aq1, b1, sa);
            const bf16_t* kp2 = Kh + (size_t)(m0 + 16 + colj) * DH;
#pragma unroll
            for (int j = 0; j < 16; ++j) { b0[j] = kp2[ks + j]; b1[j] = kp2[32 + ks + j]; }
            sb = wmma_bf16(aq0, b0, sb);  sb = wmma_bf16(aq1, b1, sb);
        }

        // ---- relative bias band: QE columns c0..c0+47 (3 tiles) ----
        // s[i,j] = q_i . E[L-1-i+j]; within a tile the band is 31 wide.
        const int c0 = L_SEQ - 16 - l0 + m0;         // >= 0; c0+47 < EPAD
        v8f qe0 = {}, qe1 = {}, qe2 = {};
#pragma unroll
        for (int t = 0; t < 3; ++t) {
            const bf16_t* ep = E + (size_t)(c0 + t * 16 + colj) * DH;
            v16bf b0, b1;
#pragma unroll
            for (int j = 0; j < 16; ++j) { b0[j] = ep[ks + j]; b1[j] = ep[32 + ks + j]; }
            v8f q = {};
            q = wmma_bf16(aq0, b0, q);  q = wmma_bf16(aq1, b1, q);
            if (t == 0) qe0 = q; else if (t == 1) qe1 = q; else qe2 = q;
        }

        // ---- add bias (shuffle along the skew diagonal), mask, online softmax
#pragma unroll
        for (int r = 0; r < 8; ++r) {
            const int ir  = r + 8 * half;            // row within tile
            const int o   = 15 - ir + colj;          // band offset, 0..30
            const int src = (lane & 16) | (o & 15);  // stay in lane-half
            const float qlo = __shfl(qe0[r], src, 32);
            const float qmi = __shfl(qe1[r], src, 32);
            const float qhi = __shfl(qe2[r], src, 32);
            float va = sa[r] + ((o < 16) ? qlo : qmi);
            float vb = sb[r] + ((o < 16) ? qmi : qhi);
            va = (m0 + colj      <= l0 + ir) ? va : -3.0e38f;  // causal
            vb = (m0 + 16 + colj <= l0 + ir) ? vb : -3.0e38f;

            float t = fmaxf(va, vb);
#pragma unroll
            for (int m = 1; m < 16; m <<= 1) t = fmaxf(t, __shfl_xor(t, m, 32));
            const float nm = fmaxf(rowmax[r], t);
            const float sc = __expf(rowmax[r] - nm);
            rowmax[r] = nm;
            const float pa = __expf(va - nm), pb = __expf(vb - nm);
            float ps = pa + pb;
#pragma unroll
            for (int m = 1; m < 16; m <<= 1) ps += __shfl_xor(ps, m, 32);
            rowsum[r] = rowsum[r] * sc + ps;
            o0[r] *= sc; o1[r] *= sc; o2[r] *= sc; o3[r] *= sc;
            pw[ir * 32 + colj]      = (bf16_t)pa;    // P in [row][k] layout
            pw[ir * 32 + 16 + colj] = (bf16_t)pb;
        }
        // wave-local LDS ops are in-order; make the stores visible to loads
        asm volatile("s_wait_dscnt 0x0" ::: "memory");

        // ---- P fragment (A-layout) from LDS ----
        v16bf pf;
        {
            const bf16_t* pp = pw + (size_t)colj * 32;
#pragma unroll
            for (int j = 0; j < 8; ++j) { pf[j] = pp[ks + j]; pf[8 + j] = pp[ks + 8 + j]; }
        }

        // ---- O += P x V (4 d-chunks of 16, K-dim = 32 keys) ----
#pragma unroll
        for (int d0 = 0; d0 < 4; ++d0) {
            const bf16_t* vp = Vh + (size_t)(d0 * 16 + colj) * L_SEQ + m0;
            v16bf bv;
#pragma unroll
            for (int j = 0; j < 16; ++j) bv[j] = vp[ks + j];
            v8f& oacc = (d0 == 0) ? o0 : (d0 == 1) ? o1 : (d0 == 2) ? o2 : o3;
            oacc = wmma_bf16(pf, bv, oacc);
        }
    }

    // ---- normalize + store merged-head bf16 output [n][l][h*64+d] ----
#pragma unroll
    for (int r = 0; r < 8; ++r) {
        const int ir = r + 8 * half;
        const float inv = 1.0f / rowsum[r];
        const size_t base = ((size_t)n * L_SEQ + (l0 + ir)) * DM + h * DH + colj;
        Ob[base +  0] = (bf16_t)(o0[r] * inv);
        Ob[base + 16] = (bf16_t)(o1[r] * inv);
        Ob[base + 32] = (bf16_t)(o2[r] * inv);
        Ob[base + 48] = (bf16_t)(o3[r] * inv);
    }
}

// ---------------------------------------------------------------------------
extern "C" void kernel_launch(void* const* d_in, const int* in_sizes, int n_in,
                              void* d_out, int out_size, void* d_ws, size_t ws_size,
                              hipStream_t stream) {
    const float* q_in = (const float*)d_in[0];
    const float* k_in = (const float*)d_in[1];
    const float* v_in = (const float*)d_in[2];
    // d_in[3] = mask (implemented analytically)
    const float* Wq = (const float*)d_in[4];
    const float* bq = (const float*)d_in[5];
    const float* Wk = (const float*)d_in[6];
    const float* bk = (const float*)d_in[7];
    const float* Wv = (const float*)d_in[8];
    const float* bv = (const float*)d_in[9];
    const float* Wo = (const float*)d_in[10];
    const float* bo = (const float*)d_in[11];
    const float* pos = (const float*)d_in[12];
    float* out = (float*)d_out;

    char* ws = (char*)d_ws;
    size_t off = 0;
    const size_t W_BYTES  = (size_t)DM * DM * sizeof(bf16_t);               // 2 MB
    const size_t HS_BYTES = (size_t)N_B * NH * L_SEQ * DH * sizeof(bf16_t); // 8 MB
    bf16_t* WqT = (bf16_t*)(ws + off); off += W_BYTES;
    bf16_t* WkT = (bf16_t*)(ws + off); off += W_BYTES;
    bf16_t* WvT = (bf16_t*)(ws + off); off += W_BYTES;
    bf16_t* WoT = (bf16_t*)(ws + off); off += W_BYTES;
    bf16_t* Qb  = (bf16_t*)(ws + off); off += HS_BYTES;
    bf16_t* Kb  = (bf16_t*)(ws + off); off += HS_BYTES;
    bf16_t* Vt  = (bf16_t*)(ws + off); off += HS_BYTES;
    bf16_t* Ob  = (bf16_t*)(ws + off); off += HS_BYTES;
    bf16_t* E   = (bf16_t*)(ws + off); off += (size_t)EPAD * DH * sizeof(bf16_t);

    // weight transpose/convert + padded E
    k_transpose_w<<<4096, 256, 0, stream>>>(Wq, WqT);
    k_transpose_w<<<4096, 256, 0, stream>>>(Wk, WkT);
    k_transpose_w<<<4096, 256, 0, stream>>>(Wv, WvT);
    k_transpose_w<<<4096, 256, 0, stream>>>(Wo, WoT);
    k_make_E<<<(EPAD * DH) / 256, 256, 0, stream>>>(pos, E);

    // projections (Q pre-scaled by 1/sqrt(depth) = 0.125); 256 blocks each
    k_gemm<float, 0><<<256, 256, 0, stream>>>(q_in, WqT, bq, Qb, 0.125f);
    k_gemm<float, 0><<<256, 256, 0, stream>>>(k_in, WkT, bk, Kb, 1.0f);
    k_gemm<float, 1><<<256, 256, 0, stream>>>(v_in, WvT, bv, Vt, 1.0f);

    // fused causal attention with relative-position bias
    k_attn<<<512, 256, 0, stream>>>(Qb, Kb, Vt, E, Ob);

    // output projection -> f32
    k_gemm<bf16_t, 2><<<256, 256, 0, stream>>>(Ob, WoT, bo, out, 1.0f);
}
